// MoeLayer_19911468384706
// MI455X (gfx1250) — matrix operand order repaired and checked by reading, not compiled, not run on hardware
//
#include <hip/hip_runtime.h>
#include <hip/hip_bf16.h>
#include <math.h>

#define NTOK 32768
#define DIM  1024
#define HID  512
#define NEXP 8

typedef __bf16 v16bf __attribute__((ext_vector_type(16)));
typedef float  v8f   __attribute__((ext_vector_type(8)));

// ---------- bf16 convert (round-to-nearest-even) ----------
__device__ __forceinline__ unsigned short f2bf(float f) {
    unsigned int u = __float_as_uint(f);
    u += 0x7FFFu + ((u >> 16) & 1u);
    return (unsigned short)(u >> 16);
}

// ================= K0: zero counters + output =================
__global__ void moe_init_kernel(int* counts, float* out) {
    const int i = blockIdx.x * 256 + threadIdx.x;
    if (i < NEXP * DIM) out[i] = 0.f;
    if (blockIdx.x == 0 && threadIdx.x < NEXP) counts[threadIdx.x] = 0;
}

// ================= K1: gating (wave per token) =================
__global__ __launch_bounds__(256) void moe_gate_kernel(
    const float* __restrict__ x, const float* __restrict__ gw,
    int* __restrict__ counts, int* __restrict__ lists) {
    const int wv = threadIdx.x >> 5;
    const int lane = threadIdx.x & 31;
    const int n = blockIdx.x * 8 + wv;

    float acc[NEXP];
#pragma unroll
    for (int e = 0; e < NEXP; ++e) acc[e] = 0.f;

    const float* xr = x + (size_t)n * DIM;
    for (int d = lane; d < DIM; d += 32) {
        float xv = xr[d];
#pragma unroll
        for (int e = 0; e < NEXP; ++e) acc[e] += xv * gw[e * DIM + d];
    }
#pragma unroll
    for (int m = 16; m >= 1; m >>= 1) {
#pragma unroll
        for (int e = 0; e < NEXP; ++e) acc[e] += __shfl_xor(acc[e], m, 32);
    }
    if (lane == 0) {
        int best = 0; float bv = acc[0];
#pragma unroll
        for (int e = 1; e < NEXP; ++e)
            if (acc[e] > bv) { bv = acc[e]; best = e; }
        int pos = atomicAdd(&counts[best], 1);
        lists[best * NTOK + pos] = n;
    }
}

// ================= K2: grouped tanh-MLP scores (WMMA bf16) =================
// Block: 256 thr (8 waves) handles 128 tokens of expert blockIdx.y.
// LDS: token ids (1 KB) | A 128x1024 bf16 (stride 2064 B) | B 1024x16 bf16 (stride 2096 B)
#define A_OFF     1024
#define A_STRIDE  2064                    // 16B aligned, conflict-free row stride
#define B_OFF     (A_OFF + 128 * A_STRIDE)
#define B_STRIDE  2096                    // 16B aligned, conflict-free col stride
#define SMEM_K2   (B_OFF + 16 * B_STRIDE) // = 298752 bytes < 320KB WGP LDS

__global__ __launch_bounds__(256) void moe_mlp_kernel(
    const float* __restrict__ x, const float* __restrict__ W1,
    const float* __restrict__ b1, const float* __restrict__ w2,
    const float* __restrict__ b2,
    const int* __restrict__ counts, const int* __restrict__ lists,
    float* __restrict__ scores) {
    extern __shared__ char smem[];
    const int e = blockIdx.y;
    const int cnt = counts[e];
    const int base = blockIdx.x * 128;
    if (base >= cnt) return;                     // uniform early exit
    const int valid = min(128, cnt - base);

    const int tid = threadIdx.x;
    int* tokv = (int*)smem;
    char* Ab = smem + A_OFF;
    char* Bb = smem + B_OFF;

    if (tid < 128) {
        int rr = base + tid;
        tokv[tid] = lists[e * NTOK + ((rr < cnt) ? rr : (cnt - 1))];
    }
    __syncthreads();

    // stage A: 128 rows x 1024 K as bf16; thread t loads float4 at k=4t of each row
    for (int r = 0; r < 128; ++r) {
        const float4 v = ((const float4*)(x + (size_t)tokv[r] * DIM))[tid];
        unsigned int lo = (unsigned int)f2bf(v.x) | ((unsigned int)f2bf(v.y) << 16);
        unsigned int hi = (unsigned int)f2bf(v.z) | ((unsigned int)f2bf(v.w) << 16);
        *(uint2*)(Ab + r * A_STRIDE + tid * 8) = make_uint2(lo, hi);
    }
    __syncthreads();

    const int wv = tid >> 5;
    const int lane = tid & 31;
    const int r0 = wv * 16;
    const char* Arow = Ab + (r0 + (lane & 15)) * A_STRIDE;
    const int aSel = (lane & 16);             // byte select within 64B K-chunk
    const char* Bcol0 = Bb + (lane & 15) * B_STRIDE;
    const int bSel = (lane & 16) ? 16 : 0;

    float sacc[8];
#pragma unroll
    for (int v = 0; v < 8; ++v) sacc[v] = 0.f;

    for (int hc = 0; hc < HID / 16; ++hc) {
        // stage B panel: W1[e][k][hc*16 + c] -> bf16 col-major [c][k]
        {
            const int c = tid & 15;
            const int kr = tid >> 4;
            const size_t wbase = ((size_t)e * DIM) * HID + (size_t)hc * 16 + c;
            for (int kk = 0; kk < DIM; kk += 16) {
                int k = kk + kr;
                float wval = W1[wbase + (size_t)k * HID];
                *(unsigned short*)(Bb + c * B_STRIDE + k * 2) = f2bf(wval);
            }
        }
        __syncthreads();

        // two independent accumulator chains to double XDL issue distance
        v8f acc0 = {0.f, 0.f, 0.f, 0.f, 0.f, 0.f, 0.f, 0.f};
        v8f acc1 = {0.f, 0.f, 0.f, 0.f, 0.f, 0.f, 0.f, 0.f};
#pragma unroll 4
        for (int ks = 0; ks < DIM / 32; ks += 2) {
            union { v16bf v; uint4 q[2]; } a0, b0, a1, b1f;
            a0.q[0] = *(const uint4*)(Arow + ks * 64 + aSel);
            a0.q[1] = *(const uint4*)(Arow + ks * 64 + aSel + 32);
            b0.q[0] = *(const uint4*)(Bcol0 + ks * 64 + bSel);
            b0.q[1] = *(const uint4*)(Bcol0 + ks * 64 + bSel + 32);
            a1.q[0] = *(const uint4*)(Arow + ks * 64 + 64 + aSel);
            a1.q[1] = *(const uint4*)(Arow + ks * 64 + 64 + aSel + 32);
            b1f.q[0] = *(const uint4*)(Bcol0 + ks * 64 + 64 + bSel);
            b1f.q[1] = *(const uint4*)(Bcol0 + ks * 64 + 64 + bSel + 32);
            acc0 = __builtin_amdgcn_wmma_f32_16x16x32_bf16(
                false, a0.v, false, b0.v, (short)0, acc0, false, false);
            acc1 = __builtin_amdgcn_wmma_f32_16x16x32_bf16(
                false, a1.v, false, b1f.v, (short)0, acc1, false, false);
        }

        const int col = hc * 16 + (lane & 15);
        const float b1v = b1[e * HID + col];
        const float w2v = w2[e * HID + col];
#pragma unroll
        for (int v = 0; v < 8; ++v)
            sacc[v] += tanhf(acc0[v] + acc1[v] + b1v) * w2v;
        __syncthreads();
    }

#pragma unroll
    for (int m = 8; m >= 1; m >>= 1) {
#pragma unroll
        for (int v = 0; v < 8; ++v) sacc[v] += __shfl_xor(sacc[v], m, 32);
    }
    if ((lane & 15) == 0) {
        const float b2v = b2[e];
        const int rbase = r0 + ((lane >> 4) << 3);
#pragma unroll
        for (int v = 0; v < 8; ++v) {
            int r = rbase + v;
            if (r < valid) scores[tokv[r]] = sacc[v] + b2v;
        }
    }
}

// ================= K3: per-expert softmax stats =================
__global__ __launch_bounds__(256) void moe_stats_kernel(
    const int* __restrict__ counts, const int* __restrict__ lists,
    const float* __restrict__ scores, float* __restrict__ emax,
    float* __restrict__ edenom) {
    __shared__ float red[256];
    const int e = blockIdx.x;
    const int cnt = counts[e];
    const int tid = threadIdx.x;

    float m = -3.4e38f;
    for (int i = tid; i < cnt; i += 256) m = fmaxf(m, scores[lists[e * NTOK + i]]);
    red[tid] = m; __syncthreads();
    for (int s = 128; s >= 1; s >>= 1) {
        if (tid < s) red[tid] = fmaxf(red[tid], red[tid + s]);
        __syncthreads();
    }
    const float gm = red[0];
    __syncthreads();

    float sum = 0.f;
    for (int i = tid; i < cnt; i += 256) sum += __expf(scores[lists[e * NTOK + i]] - gm);
    red[tid] = sum; __syncthreads();
    for (int s = 128; s >= 1; s >>= 1) {
        if (tid < s) red[tid] += red[tid + s];
        __syncthreads();
    }
    if (tid == 0) {
        emax[e] = gm;
        edenom[e] = (cnt > 0) ? red[0] : 1.0f;
    }
}

// ================= K4: attention pooling (token-sliced, atomic combine) =====
#define POOL_SLICES 16
__global__ __launch_bounds__(256) void moe_pool_kernel(
    const float* __restrict__ x, const int* __restrict__ counts,
    const int* __restrict__ lists, const float* __restrict__ scores,
    const float* __restrict__ emax, const float* __restrict__ edenom,
    float* __restrict__ out) {
    const int e = blockIdx.y;
    const int d = blockIdx.x * 256 + threadIdx.x;
    const int cnt = counts[e];
    if (cnt == 0) return;                         // out pre-zeroed
    const int per = (cnt + POOL_SLICES - 1) / POOL_SLICES;
    const int i0 = blockIdx.z * per;
    const int i1 = min(i0 + per, cnt);
    if (i0 >= i1) return;

    const float gm = emax[e];
    const float inv = 1.0f / edenom[e];
    float acc = 0.f;
    const int* lst = lists + e * NTOK;
    for (int i = i0; i < i1; ++i) {
        int t = lst[i];
        float w = __expf(scores[t] - gm) * inv;
        acc += w * x[(size_t)t * DIM + d];
    }
    atomicAdd(&out[e * DIM + d], acc);
}

// ================= host =================
extern "C" void kernel_launch(void* const* d_in, const int* in_sizes, int n_in,
                              void* d_out, int out_size, void* d_ws, size_t ws_size,
                              hipStream_t stream) {
    const float* x  = (const float*)d_in[0];   // [N, D]
    const float* gw = (const float*)d_in[1];   // [E, D]
    const float* W1 = (const float*)d_in[2];   // [E, D, H]
    const float* b1 = (const float*)d_in[3];   // [E, H]
    const float* w2 = (const float*)d_in[4];   // [E, H]
    const float* b2 = (const float*)d_in[5];   // [E]
    float* out = (float*)d_out;                // [E*D]

    char* ws = (char*)d_ws;
    int*   counts = (int*)ws;                    // E ints
    float* emax   = (float*)(ws + 64);           // E floats
    float* edenom = (float*)(ws + 128);          // E floats
    float* scores = (float*)(ws + 256);          // N floats
    int*   lists  = (int*)(ws + 256 + NTOK * 4); // E*N ints

    (void)in_sizes; (void)n_in; (void)out_size; (void)ws_size;

    hipFuncSetAttribute((const void*)moe_mlp_kernel,
                        hipFuncAttributeMaxDynamicSharedMemorySize, SMEM_K2);

    moe_init_kernel<<<(NEXP * DIM + 255) / 256, 256, 0, stream>>>(counts, out);
    moe_gate_kernel<<<NTOK / 8, 256, 0, stream>>>(x, gw, counts, lists);
    moe_mlp_kernel<<<dim3(NTOK / 128, NEXP), 256, SMEM_K2, stream>>>(
        x, W1, b1, w2, b2, counts, lists, scores);
    moe_stats_kernel<<<NEXP, 256, 0, stream>>>(counts, lists, scores, emax, edenom);
    moe_pool_kernel<<<dim3(DIM / 256, NEXP, POOL_SLICES), 256, 0, stream>>>(
        x, counts, lists, scores, emax, edenom, out);
}
